// DecoderCellPerImageWithAttention_77111842833037
// MI455X (gfx1250) — matrix operand order repaired and checked by reading, not compiled, not run on hardware
//
#include <hip/hip_runtime.h>
#include <hip/hip_bf16.h>

#define DEVI __device__ __forceinline__

typedef __attribute__((ext_vector_type(16))) _Float16 v16h;
typedef __attribute__((ext_vector_type(8)))  float    v8f;

namespace {
constexpr int nP = 640, nE = 512, nB = 32, nL = 256, nV = 8000;
constexpr int nA = 512, nD = 512, nS = 256, nED = 256, nT = 255;
constexpr int nX = nED + nE;   // 768 : LSTM input width
constexpr int nG = 4 * nD;     // 2048: LSTM gate width
}

// ---------------- WMMA fragment loaders (16-bit, contiguous k order shared by A and B) ----------------
// A fragment: 16x32 f16 tile of a row-major [M x K] matrix.
// lanes 0-15: rows m0..m0+15, K 0..15 contiguous;
// lanes 16-31: same rows, K 16..31.
DEVI v16h load_a16(const _Float16* __restrict__ A, int lda, int m0, int k0) {
  int l = threadIdx.x & 31;
  const _Float16* row = A + (size_t)(m0 + (l & 15)) * lda + k0 + ((l >> 4) << 4);
  v16h a;
#pragma unroll
  for (int v = 0; v < 8; ++v) {
    int k = 2 * v;
    a[2 * v]     = row[k];
    a[2 * v + 1] = row[k + 1];
  }
  return a;
}

// B fragment: 32x16 tile, sourced from pre-transposed weights WT [N x K] (row-major).
// lane 0-15 = column n, K 0..15 contiguous; lanes 16-31 = K 16..31. One 32B load.
DEVI v16h load_bT16(const _Float16* __restrict__ BT, int ldb, int n0, int k0) {
  const int l = threadIdx.x & 31;
  const _Float16* p = BT + (size_t)(n0 + (l & 15)) * ldb + k0 + ((l >> 4) << 4);
  return *(const v16h*)p;   // rows/offsets are 32B aligned by construction
}

DEVI v8f wmma32(const v16h& a, const v16h& b, const v8f& c) {
  return __builtin_amdgcn_wmma_f32_16x16x32_f16(false, a, false, b, (short)0, c, false, false);
}

DEVI float sigm(float x) { return 1.0f / (1.0f + __expf(-x)); }

// ---------------- setup kernels ----------------
__global__ void k_sort(const int* __restrict__ cl, const int* __restrict__ caps,
                       const float* __restrict__ hs,
                       int* __restrict__ sort_ind, int* __restrict__ dec_len,
                       int* __restrict__ caps_sorted, _Float16* __restrict__ hs16,
                       float* __restrict__ out_caps, float* __restrict__ out_declen,
                       float* __restrict__ out_sortind) {
  __shared__ int si[nB];
  if (threadIdx.x == 0) {
    for (int i = 0; i < nB; ++i) {            // stable descending argsort
      int pos = 0;
      for (int j = 0; j < nB; ++j)
        if (cl[j] > cl[i] || (cl[j] == cl[i] && j < i)) ++pos;
      si[pos] = i;
    }
    for (int r = 0; r < nB; ++r) {
      sort_ind[r] = si[r];
      int dl = cl[si[r]] - 1;
      dec_len[r] = dl;
      out_declen[r] = (float)dl;
      out_sortind[r] = (float)si[r];
    }
  }
  __syncthreads();
  for (int idx = threadIdx.x; idx < nB * nL; idx += blockDim.x) {
    int b = idx >> 8, j = idx & 255;          // nL == 256
    int v = caps[si[b] * nL + j];
    caps_sorted[idx] = v;
    out_caps[idx] = (float)v;
  }
  for (int idx = threadIdx.x; idx < nB * nS; idx += blockDim.x) {
    int b = idx >> 8, j = idx & 255;          // nS == 256
    hs16[idx] = (_Float16)hs[si[b] * nS + j];
  }
}

__global__ void k_transp16(const float* __restrict__ src, _Float16* __restrict__ dst,
                           int R, int C) {     // dst[C x R] = transpose(src[R x C]), f32->f16
  int idx = blockIdx.x * 256 + threadIdx.x;
  if (idx >= R * C) return;
  int r = idx / C, c = idx % C;
  dst[(size_t)c * R + r] = (_Float16)src[idx];
}

__global__ void k_conv16(const float* __restrict__ src, _Float16* __restrict__ dst, int n) {
  int idx = blockIdx.x * 256 + threadIdx.x;
  if (idx < n) dst[idx] = (_Float16)src[idx];
}

__global__ void k_init(const float* __restrict__ enc, const float* __restrict__ Wh0,
                       const float* __restrict__ bh0, const float* __restrict__ Wc0,
                       const float* __restrict__ bc0, float* __restrict__ hbuf,
                       float* __restrict__ cbuf, _Float16* __restrict__ h16b) {
  __shared__ float mean[nE];
  for (int e = threadIdx.x; e < nE; e += 256) {
    float s = 0.f;
    for (int p = 0; p < nP; ++p) s += enc[(size_t)p * nE + e];
    mean[e] = s * (1.0f / nP);
  }
  __syncthreads();
  for (int d = threadIdx.x; d < nD; d += 256) {
    float h = bh0[d], c = bc0[d];
    for (int e = 0; e < nE; ++e) {
      float m = mean[e];
      h = fmaf(m, Wh0[(size_t)e * nD + d], h);
      c = fmaf(m, Wc0[(size_t)e * nD + d], c);
    }
    for (int b = 0; b < nB; ++b) {            // enc identical across batch -> same h0/c0 row
      hbuf[b * nD + d] = h;
      cbuf[b * nD + d] = c;
      h16b[b * nD + d] = (_Float16)h;
    }
  }
}

// ---------------- generic WMMA GEMM: C[M x N] = A[M x K](f16) * BT[N x K](f16) ----------------
// op: 0 = bias (+ optional addm), 1 = sigmoid(bias + acc)
__global__ void k_wmma_gemm(const _Float16* __restrict__ A, int lda,
                            const _Float16* __restrict__ BT, int ldb,
                            const float* __restrict__ bias,
                            const float* __restrict__ addm,
                            float* __restrict__ C, int ldc,
                            int Mtiles, int Ntiles, int K, int op) {
  int wave = (blockIdx.x * blockDim.x + threadIdx.x) >> 5;
  if (wave >= Mtiles * Ntiles) return;
  int mt = wave % Mtiles, nt = wave / Mtiles;
  int m0 = mt * 16, n0 = nt * 16;
  v8f acc = {};
  for (int k0 = 0; k0 < K; k0 += 32)
    acc = wmma32(load_a16(A, lda, m0, k0), load_bT16(BT, ldb, n0, k0), acc);
  int l = threadIdx.x & 31;
  int n = n0 + (l & 15), mb = m0 + ((l >> 4) << 3);
#pragma unroll
  for (int r = 0; r < 8; ++r) {
    float v = acc[r];
    if (bias) v += bias[n];
    if (addm) v += addm[(size_t)(mb + r) * ldc + n];
    if (op == 1) v = sigm(v);
    C[(size_t)(mb + r) * ldc + n] = v;
  }
}

// ---------------- per-step kernels ----------------
__global__ void k_energy(const float* __restrict__ att1, const float* __restrict__ att23,
                         const float* __restrict__ Wf, const float* __restrict__ bf,
                         float* __restrict__ e) {
  int p = blockIdx.x * 8 + (threadIdx.x >> 5);   // 80 blocks x 8 rows
  int b = threadIdx.x & 31;
  const float* a1 = att1 + (size_t)p * nA;
  const float* a2 = att23 + (size_t)b * nA;
  float s = 0.f;
#pragma unroll 4
  for (int a = 0; a < nA; ++a)
    s = fmaf(fmaxf(a1[a] + a2[a], 0.f), Wf[a], s);
  e[(size_t)b * nP + p] = s + bf[0];
}

__global__ void k_softmax(const float* __restrict__ e, const int* __restrict__ dec_len,
                          int t, float* __restrict__ alphas_out,
                          _Float16* __restrict__ alpha16) {
  int b = threadIdx.x >> 5;                      // 1 block, 32 waves, wave = batch row
  int lane = threadIdx.x & 31;
  const float* row = e + (size_t)b * nP;
  float mx = -1e30f;
  for (int p = lane; p < nP; p += 32) mx = fmaxf(mx, row[p]);
  for (int o = 16; o; o >>= 1) mx = fmaxf(mx, __shfl_xor(mx, o, 32));
  float sum = 0.f;
  for (int p = lane; p < nP; p += 32) sum += __expf(row[p] - mx);
  for (int o = 16; o; o >>= 1) sum += __shfl_xor(sum, o, 32);
  float inv = 1.0f / sum;
  bool active = t < dec_len[b];
  float* arow = alphas_out + ((size_t)b * nT + t) * nP;
  _Float16* a16 = alpha16 + (size_t)b * nP;
  for (int p = lane; p < nP; p += 32) {
    float al = __expf(row[p] - mx) * inv;
    arow[p] = active ? al : 0.0f;
    a16[p] = (_Float16)al;
  }
}

__global__ void k_build_x(const int* __restrict__ caps_sorted, const float* __restrict__ emb,
                          const float* __restrict__ gate, const float* __restrict__ awe,
                          _Float16* __restrict__ x16, int t) {
  int idx = blockIdx.x * 256 + threadIdx.x;      // 96 blocks -> 32*768
  int b = idx / nX, j = idx % nX;
  float v;
  if (j < nED) {
    int tok = caps_sorted[b * nL + t];
    v = emb[(size_t)tok * nED + j];
  } else {
    int q = j - nED;
    v = gate[b * nE + q] * awe[b * nE + q];
  }
  x16[idx] = (_Float16)v;
}

__global__ void k_lstm_gates(const _Float16* __restrict__ x16, const _Float16* __restrict__ h16,
                             const _Float16* __restrict__ WihT, const _Float16* __restrict__ WhhT,
                             const float* __restrict__ bih, const float* __restrict__ bhh,
                             float* __restrict__ gates) {
  int wave = (blockIdx.x * blockDim.x + threadIdx.x) >> 5;  // 32 blocks -> 256 waves = 2x128 tiles
  int mt = wave & 1, nt = wave >> 1;
  int m0 = mt * 16, n0 = nt * 16;
  v8f acc = {};
  for (int k0 = 0; k0 < nX; k0 += 32)
    acc = wmma32(load_a16(x16, nX, m0, k0), load_bT16(WihT, nX, n0, k0), acc);
  for (int k0 = 0; k0 < nD; k0 += 32)
    acc = wmma32(load_a16(h16, nD, m0, k0), load_bT16(WhhT, nD, n0, k0), acc);
  int l = threadIdx.x & 31;
  int n = n0 + (l & 15), mb = m0 + ((l >> 4) << 3);
  float bs = bih[n] + bhh[n];
#pragma unroll
  for (int r = 0; r < 8; ++r)
    gates[(size_t)(mb + r) * nG + n] = acc[r] + bs;
}

__global__ void k_update(const float* __restrict__ gates, const float* __restrict__ c_in,
                         const float* __restrict__ h_in, float* __restrict__ c_out,
                         float* __restrict__ h_out, _Float16* __restrict__ h16_out,
                         const int* __restrict__ dec_len, int t) {
  int idx = blockIdx.x * 256 + threadIdx.x;     // 64 blocks -> 32*512
  int b = idx >> 9, d = idx & 511;
  const float* g = gates + (size_t)b * nG;
  float gi = g[d], gf = g[nD + d], gg = g[2 * nD + d], go = g[3 * nD + d];
  float cn = sigm(gf) * c_in[idx] + sigm(gi) * tanhf(gg);
  float hn = sigm(go) * tanhf(cn);
  bool active = t < dec_len[b];
  float hc = active ? hn : h_in[idx];
  float cc = active ? cn : c_in[idx];
  h_out[idx] = hc;
  c_out[idx] = cc;
  h16_out[idx] = (_Float16)hc;
}

__global__ void k_preds(const _Float16* __restrict__ h16, const _Float16* __restrict__ WoT,
                        const float* __restrict__ bo, const int* __restrict__ dec_len,
                        int t, float* __restrict__ preds) {
  int wave = (blockIdx.x * blockDim.x + threadIdx.x) >> 5;  // 125 blocks -> 1000 waves = 2x500
  int mt = wave & 1, nt = wave >> 1;
  int m0 = mt * 16, n0 = nt * 16;
  v8f acc = {};
  for (int k0 = 0; k0 < nD; k0 += 32)
    acc = wmma32(load_a16(h16, nD, m0, k0), load_bT16(WoT, nD, n0, k0), acc);
  int l = threadIdx.x & 31;
  int n = n0 + (l & 15), mb = m0 + ((l >> 4) << 3);
#pragma unroll
  for (int r = 0; r < 8; ++r) {
    int b = mb + r;
    bool active = t < dec_len[b];
    preds[((size_t)b * nT + t) * nV + n] = active ? (acc[r] + bo[n]) : 0.0f;
  }
}

// ---------------- host launch ----------------
extern "C" void kernel_launch(void* const* d_in, const int* in_sizes, int n_in,
                              void* d_out, int out_size, void* d_ws, size_t ws_size,
                              hipStream_t stream) {
  const float* enc  = (const float*)d_in[0];
  const int*   caps = (const int*)d_in[1];
  const int*   cl   = (const int*)d_in[2];
  const float* hss  = (const float*)d_in[3];
  const float* emb  = (const float*)d_in[5];
  const float* We = (const float*)d_in[6],  *be = (const float*)d_in[7];
  const float* Wd = (const float*)d_in[8],  *bd = (const float*)d_in[9];
  const float* Ws = (const float*)d_in[10], *bs = (const float*)d_in[11];
  const float* Wf = (const float*)d_in[12], *bf = (const float*)d_in[13];
  const float* Wih = (const float*)d_in[14], *bih = (const float*)d_in[15];
  const float* Whh = (const float*)d_in[16], *bhh = (const float*)d_in[17];
  const float* Wh0 = (const float*)d_in[18], *bh0 = (const float*)d_in[19];
  const float* Wc0 = (const float*)d_in[20], *bc0 = (const float*)d_in[21];
  const float* Wg = (const float*)d_in[22], *bg = (const float*)d_in[23];
  const float* Wo = (const float*)d_in[24], *bo = (const float*)d_in[25];

  float* out = (float*)d_out;
  const size_t OFF_PREDS  = 0;
  const size_t OFF_CAPS   = (size_t)nB * nT * nV;
  const size_t OFF_DECLEN = OFF_CAPS + (size_t)nB * nL;
  const size_t OFF_ALPHA  = OFF_DECLEN + nB;
  const size_t OFF_SORT   = OFF_ALPHA + (size_t)nB * nT * nP;

  char* w = (char*)d_ws;
  auto alloc = [&](size_t bytes) -> void* {
    void* p = (void*)w;
    w += (bytes + 255) & ~(size_t)255;
    return p;
  };
  _Float16* WdT  = (_Float16*)alloc((size_t)nA * nD * 2);
  _Float16* WgT  = (_Float16*)alloc((size_t)nE * nD * 2);
  _Float16* WihT = (_Float16*)alloc((size_t)nG * nX * 2);
  _Float16* WhhT = (_Float16*)alloc((size_t)nG * nD * 2);
  _Float16* WoT  = (_Float16*)alloc((size_t)nV * nD * 2);
  _Float16* WeT  = (_Float16*)alloc((size_t)nA * nE * 2);
  _Float16* WsT  = (_Float16*)alloc((size_t)nA * nS * 2);
  _Float16* enc16  = (_Float16*)alloc((size_t)nP * nE * 2);
  _Float16* encT16 = (_Float16*)alloc((size_t)nE * nP * 2);
  float* att1  = (float*)alloc((size_t)nP * nA * 4);
  float* att3  = (float*)alloc((size_t)nB * nA * 4);
  _Float16* hs16 = (_Float16*)alloc((size_t)nB * nS * 2);
  float* hbuf[2]  = {(float*)alloc((size_t)nB * nD * 4), (float*)alloc((size_t)nB * nD * 4)};
  float* cbuf[2]  = {(float*)alloc((size_t)nB * nD * 4), (float*)alloc((size_t)nB * nD * 4)};
  _Float16* h16b[2] = {(_Float16*)alloc((size_t)nB * nD * 2), (_Float16*)alloc((size_t)nB * nD * 2)};
  float* att23 = (float*)alloc((size_t)nB * nA * 4);
  float* gate  = (float*)alloc((size_t)nB * nE * 4);
  float* ebuf  = (float*)alloc((size_t)nB * nP * 4);
  _Float16* alpha16 = (_Float16*)alloc((size_t)nB * nP * 2);
  float* awe   = (float*)alloc((size_t)nB * nE * 4);
  _Float16* x16 = (_Float16*)alloc((size_t)nB * nX * 2);
  float* gates = (float*)alloc((size_t)nB * nG * 4);
  int* caps_sorted = (int*)alloc((size_t)nB * nL * 4);
  int* dec_len = (int*)alloc(nB * 4);
  int* sort_ind = (int*)alloc(nB * 4);

  // ---- setup ----
  k_sort<<<1, 256, 0, stream>>>(cl, caps, hss, sort_ind, dec_len, caps_sorted, hs16,
                                out + OFF_CAPS, out + OFF_DECLEN, out + OFF_SORT);
  auto transp = [&](const float* s, _Float16* d, int R, int C) {
    k_transp16<<<(R * C + 255) / 256, 256, 0, stream>>>(s, d, R, C);
  };
  transp(Wd, WdT, nD, nA);
  transp(Wg, WgT, nD, nE);
  transp(Wih, WihT, nX, nG);
  transp(Whh, WhhT, nD, nG);
  transp(Wo, WoT, nD, nV);
  transp(We, WeT, nE, nA);
  transp(Ws, WsT, nS, nA);
  transp(enc, encT16, nP, nE);
  k_conv16<<<(nP * nE + 255) / 256, 256, 0, stream>>>(enc, enc16, nP * nE);
  k_init<<<1, 256, 0, stream>>>(enc, Wh0, bh0, Wc0, bc0, hbuf[0], cbuf[0], h16b[0]);
  // att1 = enc @ We + be : 40x32 tiles
  k_wmma_gemm<<<160, 256, 0, stream>>>(enc16, nE, WeT, nE, be, nullptr, att1, nA, 40, 32, nE, 0);
  // att3 = hs_sorted @ Ws + bs : 2x32 tiles
  k_wmma_gemm<<<8, 256, 0, stream>>>(hs16, nS, WsT, nS, bs, nullptr, att3, nA, 2, 32, nS, 0);

  // ---- recurrence ----
  for (int t = 0; t < nT; ++t) {
    int cur = t & 1, nxt = cur ^ 1;
    // att23 = h@Wd + bd + att3 ; gate = sigmoid(h@Wg + bg)
    k_wmma_gemm<<<8, 256, 0, stream>>>(h16b[cur], nD, WdT, nD, bd, att3, att23, nA, 2, 32, nD, 0);
    k_wmma_gemm<<<8, 256, 0, stream>>>(h16b[cur], nD, WgT, nD, bg, nullptr, gate, nE, 2, 32, nD, 1);
    k_energy<<<nP / 8, 256, 0, stream>>>(att1, att23, Wf, bf, ebuf);
    k_softmax<<<1, 1024, 0, stream>>>(ebuf, dec_len, t, out + OFF_ALPHA, alpha16);
    // awe = alpha @ enc
    k_wmma_gemm<<<8, 256, 0, stream>>>(alpha16, nP, encT16, nP, nullptr, nullptr, awe, nE, 2, 32, nP, 0);
    k_build_x<<<(nB * nX) / 256, 256, 0, stream>>>(caps_sorted, emb, gate, awe, x16, t);
    k_lstm_gates<<<32, 256, 0, stream>>>(x16, h16b[cur], WihT, WhhT, bih, bhh, gates);
    k_update<<<(nB * nD) / 256, 256, 0, stream>>>(gates, cbuf[cur], hbuf[cur],
                                                  cbuf[nxt], hbuf[nxt], h16b[nxt], dec_len, t);
    k_preds<<<125, 256, 0, stream>>>(h16b[nxt], WoT, bo, dec_len, t, out + OFF_PREDS);
  }
  (void)in_sizes; (void)n_in; (void)out_size; (void)ws_size;
}
